// TransformerBlock_89575837925711
// MI455X (gfx1250) — matrix operand order repaired and checked by reading, not compiled, not run on hardware
//
#include <hip/hip_runtime.h>
#include <cmath>

typedef _Float16 half8 __attribute__((ext_vector_type(8)));
typedef _Float16 v16h  __attribute__((ext_vector_type(16)));
typedef float    v8f   __attribute__((ext_vector_type(8)));

#define BB  2
#define TT  2048
#define CC  1024
#define HH  16
#define HDD 64
#define MM  (BB * TT)   // 4096 rows

static __device__ __forceinline__ v16h cat8(half8 lo, half8 hi) {
  return __builtin_shufflevector(lo, hi, 0,1,2,3,4,5,6,7,8,9,10,11,12,13,14,15);
}
static __device__ __forceinline__ v8f zero8() {
  v8f z = {0.f,0.f,0.f,0.f,0.f,0.f,0.f,0.f};
  return z;
}
static __device__ __forceinline__ float gelu_exact(float x) {
  return 0.5f * x * (1.0f + erff(x * 0.70710678118654752f));
}

// ---------------------------------------------------------------------------
// Weight convert + transpose: w (K x N, f32) -> wt (N x K, f16)
// ---------------------------------------------------------------------------
__global__ __launch_bounds__(256) void convert_wT_kernel(
    const float* __restrict__ w, _Float16* __restrict__ wt, int K, int N) {
  size_t idx = (size_t)blockIdx.x * 256 + threadIdx.x;
  if (idx >= (size_t)K * N) return;
  int k = (int)(idx / N);
  int n = (int)(idx % N);
  wt[(size_t)n * K + k] = (_Float16)w[idx];
}

// ---------------------------------------------------------------------------
// LayerNorm: per-row over C; optional fp32 output + f16 output
// ---------------------------------------------------------------------------
__global__ __launch_bounds__(256) void ln_kernel(
    const float* __restrict__ x, const float* __restrict__ g,
    const float* __restrict__ bta, float* __restrict__ y32,
    _Float16* __restrict__ y16, int C) {
  __shared__ float red[256];
  const int row = blockIdx.x;
  const int tid = threadIdx.x;
  const float* xr = x + (size_t)row * C;
  float s = 0.f, s2 = 0.f;
  for (int i = tid; i < C; i += 256) { float v = xr[i]; s += v; s2 += v * v; }
  red[tid] = s; __syncthreads();
  for (int st = 128; st > 0; st >>= 1) {
    if (tid < st) red[tid] += red[tid + st];
    __syncthreads();
  }
  float mean = red[0] / (float)C;
  __syncthreads();
  red[tid] = s2; __syncthreads();
  for (int st = 128; st > 0; st >>= 1) {
    if (tid < st) red[tid] += red[tid + st];
    __syncthreads();
  }
  float var = red[0] / (float)C - mean * mean;
  float inv = rsqrtf(var + 1e-5f);
  for (int i = tid; i < C; i += 256) {
    float v = (xr[i] - mean) * inv * g[i] + bta[i];
    if (y32) y32[(size_t)row * C + i] = v;
    y16[(size_t)row * C + i] = (_Float16)v;
  }
}

// ---------------------------------------------------------------------------
// Tiled f16 GEMM with f32 WMMA accumulate, double-buffered LDS pipeline.
//   C[M,N] = A[M,K] @ BT[N,K]^T + bias;  optional GELU, residual, f32/f16 out
// Block: 256 threads (8 waves), tile 128x128, K-step 32, 1 barrier/step.
// Wave grid 2(M) x 4(N); each wave: 4x2 subtiles of 16x16.
// ---------------------------------------------------------------------------
#define GLDA 40   // 32 + 8 halfs pad; 80B row stride (16B aligned)
__global__ __launch_bounds__(256) void gemm_f16_kernel(
    const _Float16* __restrict__ A, const _Float16* __restrict__ BT,
    const float* __restrict__ bias, const float* __restrict__ resid,
    float* __restrict__ Cf, _Float16* __restrict__ Ch,
    int M, int N, int K, int act) {
  __shared__ __attribute__((aligned(16))) _Float16 As[2][128 * GLDA];
  __shared__ __attribute__((aligned(16))) _Float16 Bs[2][128 * GLDA];
  const int tid  = threadIdx.x;
  const int lane = tid & 31, wave = tid >> 5;
  const int llo  = lane & 15, lhi = lane >> 4;
  const int wm   = wave & 1,  wn  = wave >> 1;
  const int tm   = blockIdx.y * 128, tn = blockIdx.x * 128;
  const int ldrow = tid >> 2;            // 0..63 (+64 on 2nd slot)
  const int ldch  = (tid & 3) * 8;       // 0,8,16,24

  v8f acc[4][2];
#pragma unroll
  for (int i = 0; i < 4; ++i)
#pragma unroll
    for (int j = 0; j < 2; ++j) acc[i][j] = zero8();

  const int KT = K / 32;
  half8 areg[2], breg[2];

  // prologue: stage tile 0 and commit to buffer 0
#pragma unroll
  for (int it = 0; it < 2; ++it) {
    int row = ldrow + it * 64;
    areg[it] = *(const half8*)&A[(size_t)(tm + row) * K + ldch];
    breg[it] = *(const half8*)&BT[(size_t)(tn + row) * K + ldch];
  }
#pragma unroll
  for (int it = 0; it < 2; ++it) {
    int row = ldrow + it * 64;
    *(half8*)&As[0][row * GLDA + ldch] = areg[it];
    *(half8*)&Bs[0][row * GLDA + ldch] = breg[it];
  }

  for (int kt = 0; kt < KT; ++kt) {
    __syncthreads();
    // issue next tile's global loads before the WMMA burst (latency hiding)
    if (kt + 1 < KT) {
      int k0n = (kt + 1) * 32;
#pragma unroll
      for (int it = 0; it < 2; ++it) {
        int row = ldrow + it * 64;
        areg[it] = *(const half8*)&A[(size_t)(tm + row) * K + k0n + ldch];
        breg[it] = *(const half8*)&BT[(size_t)(tn + row) * K + k0n + ldch];
      }
    }
    if (kt + 2 < KT) {   // warm L2 two tiles ahead (global_prefetch_b8)
      int k0p = (kt + 2) * 32;
      __builtin_prefetch(&A[(size_t)(tm + ldrow) * K + k0p + ldch], 0, 1);
      __builtin_prefetch(&BT[(size_t)(tn + ldrow) * K + k0p + ldch], 0, 1);
    }

    const _Float16* Ab = As[kt & 1];
    const _Float16* Bb = Bs[kt & 1];
    v16h afr[4], bfr[2];
#pragma unroll
    for (int sm = 0; sm < 4; ++sm) {
      int r = wm * 64 + sm * 16 + llo;
      half8 lo = *(const half8*)&Ab[r * GLDA + 8 * lhi];
      half8 hi = *(const half8*)&Ab[r * GLDA + 16 + 8 * lhi];
      afr[sm] = cat8(lo, hi);
    }
#pragma unroll
    for (int sn = 0; sn < 2; ++sn) {
      int c = wn * 32 + sn * 16 + llo;
      half8 lo = *(const half8*)&Bb[c * GLDA + 8 * lhi];
      half8 hi = *(const half8*)&Bb[c * GLDA + 16 + 8 * lhi];
      bfr[sn] = cat8(lo, hi);
    }
#pragma unroll
    for (int sm = 0; sm < 4; ++sm)
#pragma unroll
      for (int sn = 0; sn < 2; ++sn)
        acc[sm][sn] = __builtin_amdgcn_wmma_f32_16x16x32_f16(
            false, afr[sm], false, bfr[sn], (short)0, acc[sm][sn], false, false);

    // commit staged tile into the other buffer (safe: last reads of that
    // buffer were before the barrier at the top of this iteration)
    if (kt + 1 < KT) {
      int nb = (kt + 1) & 1;
#pragma unroll
      for (int it = 0; it < 2; ++it) {
        int row = ldrow + it * 64;
        *(half8*)&As[nb][row * GLDA + ldch] = areg[it];
        *(half8*)&Bs[nb][row * GLDA + ldch] = breg[it];
      }
    }
  }

#pragma unroll
  for (int sm = 0; sm < 4; ++sm)
#pragma unroll
    for (int sn = 0; sn < 2; ++sn)
#pragma unroll
      for (int r = 0; r < 8; ++r) {
        int rg = tm + wm * 64 + sm * 16 + r + 8 * lhi;
        int cg = tn + wn * 32 + sn * 16 + llo;
        float v = acc[sm][sn][r] + bias[cg];
        if (act) v = gelu_exact(v);
        if (resid) v += resid[(size_t)rg * N + cg];
        if (Cf) Cf[(size_t)rg * N + cg] = v;
        if (Ch) Ch[(size_t)rg * N + cg] = (_Float16)v;
      }
}

// ---------------------------------------------------------------------------
// QKV repack: qkv16[M,3C] -> Q/K [bh][t][d] f16, Vt [bh][d][t] f16
// ---------------------------------------------------------------------------
__global__ __launch_bounds__(256) void repack_qkv_kernel(
    const _Float16* __restrict__ qkv, _Float16* __restrict__ q,
    _Float16* __restrict__ k, _Float16* __restrict__ vt) {
  size_t idx = (size_t)blockIdx.x * 256 + threadIdx.x;  // over B*T*H*HD
  if (idx >= (size_t)MM * CC) return;
  int d   = (int)(idx & 63);
  size_t tmp = idx >> 6;
  int hh  = (int)(tmp & 15);
  size_t bt = tmp >> 4;                // b*T + t
  int t   = (int)(bt % TT);
  int b   = (int)(bt / TT);
  const _Float16* src = qkv + bt * (3 * CC) + hh * (3 * HDD);
  size_t bh = (size_t)b * HH + hh;
  q [(bh * TT + t) * HDD + d] = src[d];
  k [(bh * TT + t) * HDD + d] = src[HDD + d];
  vt[(bh * HDD + d) * TT + t] = src[2 * HDD + d];
}

// ---------------------------------------------------------------------------
// Flash attention (causal). Block = (qt, bh), 128 threads = 4 waves.
// Each wave owns 16 query rows; S tile 16x64 per wave via WMMA; online softmax.
// K/V tiles double-buffered in LDS, one barrier per key tile.
// ---------------------------------------------------------------------------
#define VLDA 72   // 64 + 8 halfs pad; 144B row stride (16B aligned)
__global__ __launch_bounds__(128) void attn_kernel(
    const _Float16* __restrict__ Q, const _Float16* __restrict__ Kc,
    const _Float16* __restrict__ Vt, _Float16* __restrict__ O) {
  __shared__ __attribute__((aligned(16))) _Float16 Qs[64 * VLDA];
  __shared__ __attribute__((aligned(16))) _Float16 Ks[2][64 * VLDA];
  __shared__ __attribute__((aligned(16))) _Float16 Vs[2][64 * VLDA];
  __shared__ __attribute__((aligned(16))) _Float16 Ps[4][16 * VLDA];

  const int tid  = threadIdx.x;
  const int lane = tid & 31, wave = tid >> 5;
  const int llo  = lane & 15, lhi = lane >> 4;
  const int qt   = blockIdx.x;     // query tile (64 rows)
  const int bh   = blockIdx.y;     // b*H + h
  const size_t qbase = (size_t)bh * TT * HDD;
  const size_t vbase = (size_t)bh * HDD * TT;
  const float NEGINF = -3.0e38f;
  const int ldrow = tid >> 3;      // 0..15 (+16/32/48 over 4 slots)
  const int ldch  = (tid & 7) * 8; // 0..56

  // load Q tile (64 x 64)
#pragma unroll
  for (int it = 0; it < 4; ++it) {
    int row = ldrow + it * 16;
    *(half8*)&Qs[row * VLDA + ldch] =
        *(const half8*)&Q[qbase + (size_t)(qt * 64 + row) * HDD + ldch];
  }

  // prologue: stage key tile 0 and commit to buffer 0
  half8 kreg[4], vreg[4];
#pragma unroll
  for (int it = 0; it < 4; ++it) {
    int row = ldrow + it * 16;
    kreg[it] = *(const half8*)&Kc[qbase + (size_t)row * HDD + ldch];
    vreg[it] = *(const half8*)&Vt[vbase + (size_t)row * TT + ldch];
  }
#pragma unroll
  for (int it = 0; it < 4; ++it) {
    int row = ldrow + it * 16;
    *(half8*)&Ks[0][row * VLDA + ldch] = kreg[it];
    *(half8*)&Vs[0][row * VLDA + ldch] = vreg[it];
  }

  v8f accO[4];
  float mrow[8], lrow[8];
#pragma unroll
  for (int sn = 0; sn < 4; ++sn) accO[sn] = zero8();
#pragma unroll
  for (int r = 0; r < 8; ++r) { mrow[r] = NEGINF; lrow[r] = 0.f; }

  for (int kt = 0; kt <= qt; ++kt) {
    __syncthreads();
    // issue next key tile's global loads before the compute burst
    if (kt < qt) {
#pragma unroll
      for (int it = 0; it < 4; ++it) {
        int row = ldrow + it * 16;
        kreg[it] = *(const half8*)&Kc[qbase + (size_t)((kt + 1) * 64 + row) * HDD + ldch];
        vreg[it] = *(const half8*)&Vt[vbase + (size_t)row * TT + (kt + 1) * 64 + ldch];
      }
    }
    const _Float16* Kb = Ks[kt & 1];
    const _Float16* Vb = Vs[kt & 1];

    // S = Q @ K^T (16 q-rows x 64 keys per wave)
    v8f accS[4];
#pragma unroll
    for (int sn = 0; sn < 4; ++sn) accS[sn] = zero8();
#pragma unroll
    for (int c = 0; c < 2; ++c) {
      int qrow = wave * 16 + llo;
      half8 alo = *(const half8*)&Qs[qrow * VLDA + 32 * c + 8 * lhi];
      half8 ahi = *(const half8*)&Qs[qrow * VLDA + 32 * c + 16 + 8 * lhi];
      v16h af = cat8(alo, ahi);
#pragma unroll
      for (int sn = 0; sn < 4; ++sn) {
        int krow = sn * 16 + llo;
        half8 blo = *(const half8*)&Kb[krow * VLDA + 32 * c + 8 * lhi];
        half8 bhi = *(const half8*)&Kb[krow * VLDA + 32 * c + 16 + 8 * lhi];
        accS[sn] = __builtin_amdgcn_wmma_f32_16x16x32_f16(
            false, af, false, cat8(blo, bhi), (short)0, accS[sn], false, false);
      }
    }

    // scale + causal mask + row max (elem r -> row r+8*lhi, col -> llo)
    float mnew[8];
#pragma unroll
    for (int r = 0; r < 8; ++r) {
      float mx = NEGINF;
#pragma unroll
      for (int sn = 0; sn < 4; ++sn) {
        float s = accS[sn][r] * 0.125f;   // 1/sqrt(64)
        if (kt == qt && (sn * 16 + llo) > (wave * 16 + r + 8 * lhi)) s = NEGINF;
        accS[sn][r] = s;
        mx = fmaxf(mx, s);
      }
#pragma unroll
      for (int o2 = 1; o2 < 16; o2 <<= 1) mx = fmaxf(mx, __shfl_xor(mx, o2, 32));
      mnew[r] = fmaxf(mrow[r], mx);
    }

    // p = exp(s - mnew), row sums, rescale running O and l
#pragma unroll
    for (int r = 0; r < 8; ++r) {
      float scl = expf(mrow[r] - mnew[r]);
      float rs = 0.f;
#pragma unroll
      for (int sn = 0; sn < 4; ++sn) {
        float p = expf(accS[sn][r] - mnew[r]);
        accS[sn][r] = p;
        rs += p;
      }
#pragma unroll
      for (int o2 = 1; o2 < 16; o2 <<= 1) rs += __shfl_xor(rs, o2, 32);
      lrow[r] = lrow[r] * scl + rs;
      mrow[r] = mnew[r];
#pragma unroll
      for (int sn = 0; sn < 4; ++sn) accO[sn][r] *= scl;
    }

    // stage P (C-layout -> A-layout) through per-wave LDS
#pragma unroll
    for (int sn = 0; sn < 4; ++sn)
#pragma unroll
      for (int r = 0; r < 8; ++r)
        Ps[wave][(r + 8 * lhi) * VLDA + sn * 16 + llo] = (_Float16)accS[sn][r];

    // O += P @ V   (A = P[16 x 64 keys], B = Vt[d][key])
#pragma unroll
    for (int c = 0; c < 2; ++c) {
      half8 alo = *(const half8*)&Ps[wave][llo * VLDA + 32 * c + 8 * lhi];
      half8 ahi = *(const half8*)&Ps[wave][llo * VLDA + 32 * c + 16 + 8 * lhi];
      v16h af = cat8(alo, ahi);
#pragma unroll
      for (int sn = 0; sn < 4; ++sn) {
        int vrow = sn * 16 + llo;
        half8 blo = *(const half8*)&Vb[vrow * VLDA + 32 * c + 8 * lhi];
        half8 bhi = *(const half8*)&Vb[vrow * VLDA + 32 * c + 16 + 8 * lhi];
        accO[sn] = __builtin_amdgcn_wmma_f32_16x16x32_f16(
            false, af, false, cat8(blo, bhi), (short)0, accO[sn], false, false);
      }
    }

    // commit staged next K/V tile (other buffer; safe past top barrier)
    if (kt < qt) {
      int nb = (kt + 1) & 1;
#pragma unroll
      for (int it = 0; it < 4; ++it) {
        int row = ldrow + it * 16;
        *(half8*)&Ks[nb][row * VLDA + ldch] = kreg[it];
        *(half8*)&Vs[nb][row * VLDA + ldch] = vreg[it];
      }
    }
  }

  // epilogue: O /= l, write f16 at [b][t][h*64+d]
  const int bq = bh >> 4;       // / HH
  const int hh = bh & 15;       // % HH
#pragma unroll
  for (int r = 0; r < 8; ++r) {
    float inv = 1.0f / lrow[r];
    int t = qt * 64 + wave * 16 + r + 8 * lhi;
#pragma unroll
    for (int sn = 0; sn < 4; ++sn) {
      int d = sn * 16 + llo;
      O[((size_t)bq * TT + t) * CC + hh * HDD + d] = (_Float16)(accO[sn][r] * inv);
    }
  }
}

// ---------------------------------------------------------------------------
// Host-side orchestration
// ---------------------------------------------------------------------------
extern "C" void kernel_launch(void* const* d_in, const int* in_sizes, int n_in,
                              void* d_out, int out_size, void* d_ws, size_t ws_size,
                              hipStream_t stream) {
  const float* x      = (const float*)d_in[0];
  const float* ln1_g  = (const float*)d_in[1];
  const float* ln1_b  = (const float*)d_in[2];
  const float* ln2_g  = (const float*)d_in[3];
  const float* ln2_b  = (const float*)d_in[4];
  const float* w_in   = (const float*)d_in[5];
  const float* b_in   = (const float*)d_in[6];
  const float* w_out  = (const float*)d_in[7];
  const float* b_out  = (const float*)d_in[8];
  const float* w_fc   = (const float*)d_in[9];
  const float* b_fc   = (const float*)d_in[10];
  const float* w_proj = (const float*)d_in[11];
  const float* b_proj = (const float*)d_in[12];
  float* out = (float*)d_out;

  char* ws = (char*)d_ws;
  size_t off = 0;
  auto take = [&](size_t bytes) {
    void* p = ws + off;
    off = (off + bytes + 255) & ~(size_t)255;
    return p;
  };

  _Float16* wT_in   = (_Float16*)take((size_t)3 * CC * CC * 2);
  _Float16* wT_out  = (_Float16*)take((size_t)CC * CC * 2);
  _Float16* wT_fc   = (_Float16*)take((size_t)4 * CC * CC * 2);
  _Float16* wT_proj = (_Float16*)take((size_t)4 * CC * CC * 2);
  _Float16* h16     = (_Float16*)take((size_t)MM * CC * 2);
  _Float16* qkv16   = (_Float16*)take((size_t)MM * 3 * CC * 2);
  _Float16* q16     = (_Float16*)take((size_t)MM * CC * 2);
  _Float16* k16     = (_Float16*)take((size_t)MM * CC * 2);
  _Float16* vt16    = (_Float16*)take((size_t)MM * CC * 2);
  _Float16* o16     = (_Float16*)take((size_t)MM * CC * 2);
  float*    x1      = (float*)take((size_t)MM * CC * 4);
  float*    y32     = (float*)take((size_t)MM * CC * 4);
  _Float16* y16     = (_Float16*)take((size_t)MM * CC * 2);
  _Float16* u16     = (_Float16*)take((size_t)MM * 4 * CC * 2);
  (void)ws_size; (void)in_sizes; (void)n_in; (void)out_size;

  // weight transposes (f32 -> f16 [N][K])
  convert_wT_kernel<<<(3 * CC * CC + 255) / 256, 256, 0, stream>>>(w_in,   wT_in,   CC,     3 * CC);
  convert_wT_kernel<<<(CC * CC + 255) / 256,     256, 0, stream>>>(w_out,  wT_out,  CC,     CC);
  convert_wT_kernel<<<(4 * CC * CC + 255) / 256, 256, 0, stream>>>(w_fc,   wT_fc,   CC,     4 * CC);
  convert_wT_kernel<<<(4 * CC * CC + 255) / 256, 256, 0, stream>>>(w_proj, wT_proj, 4 * CC, CC);

  // h = ln1(x)  (f16 only)
  ln_kernel<<<MM, 256, 0, stream>>>(x, ln1_g, ln1_b, nullptr, h16, CC);

  // qkv = h @ w_in + b_in   (f16 out)
  gemm_f16_kernel<<<dim3(3 * CC / 128, MM / 128), 256, 0, stream>>>(
      h16, wT_in, b_in, nullptr, nullptr, qkv16, MM, 3 * CC, CC, 0);

  repack_qkv_kernel<<<(MM * CC + 255) / 256, 256, 0, stream>>>(qkv16, q16, k16, vt16);

  // causal flash attention -> o16 [b][t][h*64+d]
  attn_kernel<<<dim3(TT / 64, BB * HH), 128, 0, stream>>>(q16, k16, vt16, o16);

  // x1 = x + o @ w_out + b_out   (f32 out)
  gemm_f16_kernel<<<dim3(CC / 128, MM / 128), 256, 0, stream>>>(
      o16, wT_out, b_out, x, x1, nullptr, MM, CC, CC, 0);

  // y = ln2(x1)  (f32 + f16)
  ln_kernel<<<MM, 256, 0, stream>>>(x1, ln2_g, ln2_b, y32, y16, CC);

  // u = gelu(y @ w_fc + b_fc)   (f16 out)
  gemm_f16_kernel<<<dim3(4 * CC / 128, MM / 128), 256, 0, stream>>>(
      y16, wT_fc, b_fc, nullptr, nullptr, u16, MM, 4 * CC, CC, 1);

  // out = y + u @ w_proj + b_proj   (f32 out)
  gemm_f16_kernel<<<dim3(CC / 128, MM / 128), 256, 0, stream>>>(
      u16, wT_proj, b_proj, y32, out, nullptr, MM, CC, 4 * CC, 0);
}
